// MultiHeadAttention_81338090651691
// MI455X (gfx1250) — matrix operand order repaired and checked by reading, not compiled, not run on hardware
//
#include <hip/hip_runtime.h>

typedef __bf16 bf16_t;
typedef __attribute__((ext_vector_type(16))) __bf16 v16bf;
typedef __attribute__((ext_vector_type(8)))  __bf16 bf16x8;
typedef __attribute__((ext_vector_type(8)))  float  v8f;
typedef __attribute__((ext_vector_type(4)))  unsigned int u32x4;
typedef __attribute__((ext_vector_type(4)))  int i32x4;
typedef __attribute__((ext_vector_type(8)))  int i32x8;

#define SEQ_T 2048
#define DK    128
#define NH    16
#define NB    2

// ---------- WMMA fragment loaders (wave32, bf16 16x16x32) ----------
// A 16x32: lane holds row m=lane&15; 8 contiguous bf16 at k0=8*(lane>>4), plus same at +16.
__device__ __forceinline__ v16bf load_fragA(const bf16_t* p) {
  bf16x8 a = *(const bf16x8*)(p);
  bf16x8 b = *(const bf16x8*)(p + 16);
  v16bf r;
#pragma unroll
  for (int i = 0; i < 8; ++i) { r[i] = a[i]; r[i + 8] = b[i]; }
  return r;
}
// B 32x16: lane holds col n=lane&15; 8 contiguous bf16 at k0=8*(lane>>4), plus same at +16.
__device__ __forceinline__ v16bf load_fragB(const bf16_t* p) {
  bf16x8 a = *(const bf16x8*)(p);
  bf16x8 b = *(const bf16x8*)(p + 16);
  v16bf r;
#pragma unroll
  for (int i = 0; i < 8; ++i) { r[i] = a[i]; r[i + 8] = b[i]; }
  return r;
}

// ---------- Tensor Data Mover: async 2D tile load global -> LDS ----------
// D# per CDNA5 ISA ch.8: group0 = {count, lds_addr, global_addr, type}, group1 =
// {data_size, tensor dims (huge => no OOB clipping), tile dims, dim0 stride}.
__device__ __forceinline__ void tdm_load_tile_2d(void* ldsDst, const void* gsrc,
                                                 unsigned tileD0, unsigned tileD1,
                                                 unsigned long long strideElems) {
#if defined(__HIP_DEVICE_COMPILE__)
  const unsigned long long ga = (unsigned long long)gsrc;
  const unsigned lds = (unsigned)(unsigned long long)ldsDst;   // LDS byte offset (low bits)
  u32x4 g0;
  g0.x = 1u;                                                   // count=1, user descriptor
  g0.y = lds;                                                  // lds_addr
  g0.z = (unsigned)ga;                                         // global_addr[31:0]
  g0.w = (unsigned)((ga >> 32) & 0x01FFFFFFu) | (2u << 30);    // global_addr[56:32], type=2
  const unsigned td0 = 1u << 20, td1 = 1u << 20;               // tensor dims >> tile: no OOB
  i32x8 g1;
  g1[0] = (int)(1u << 16);                                     // data_size=1 (2B), no multicast
  g1[1] = (int)((td0 & 0xFFFFu) << 16);                        // tensor_dim0[15:0]
  g1[2] = (int)(((td0 >> 16) & 0xFFFFu) | ((td1 & 0xFFFFu) << 16));
  g1[3] = (int)(((td1 >> 16) & 0xFFFFu) | ((tileD0 & 0xFFFFu) << 16));
  g1[4] = (int)(tileD1 & 0xFFFFu);                             // tile_dim1; tile_dim2=0
  g1[5] = (int)(unsigned)(strideElems & 0xFFFFFFFFull);        // tensor_dim0_stride[31:0]
  g1[6] = (int)(unsigned)((strideElems >> 32) & 0xFFFFull);    // stride[47:32]
  g1[7] = 0;
  i32x4 z4 = {0, 0, 0, 0};
#if defined(__clang_major__) && (__clang_major__ >= 23)
  i32x8 z8 = {0, 0, 0, 0, 0, 0, 0, 0};
  __builtin_amdgcn_tensor_load_to_lds(g0, g1, z4, z4, z8, 0);
#else
  __builtin_amdgcn_tensor_load_to_lds(g0, g1, z4, z4, 0);
#endif
#endif
}

// s_wait_tensorcnt requires an integer-constant operand -> template parameter.
template <int N>
__device__ __forceinline__ void tdm_wait() {
#if defined(__HIP_DEVICE_COMPILE__)
  __builtin_amdgcn_s_wait_tensorcnt(N);
#endif
}

// ---------- fp32 -> bf16 conversion ----------
__global__ void mha_f32_to_bf16_k(const float* __restrict__ in, bf16_t* __restrict__ out, int n) {
  int i = blockIdx.x * blockDim.x + threadIdx.x;
  int stride = gridDim.x * blockDim.x;
  for (; i < n; i += stride) out[i] = (bf16_t)in[i];
}

// ---------- GEMM: C[M,N] = A[M,K] * B[N,K]^T, bf16 in, fp32 out ----------
// 256 threads = 8 waves; block tile 128x256; wave tile 64x64 (4x4 WMMA tiles).
__global__ __launch_bounds__(256) void mha_gemm_bf16_wmma_k(
    const bf16_t* __restrict__ A, const bf16_t* __restrict__ B,
    float* __restrict__ C, int M, int N, int K) {
  const int lane = threadIdx.x & 31;
  const int w    = threadIdx.x >> 5;
  const int lo   = lane & 15;
  const int hi   = lane >> 4;
  const int rowBase = blockIdx.y * 128 + (w >> 2) * 64;
  const int colBase = blockIdx.x * 256 + (w & 3) * 64;

  v8f acc[4][4];
#pragma unroll
  for (int i = 0; i < 4; ++i)
#pragma unroll
    for (int j = 0; j < 4; ++j) acc[i][j] = {};

  const bf16_t* aBase = A + (size_t)(rowBase + lo) * K + 8 * hi;
  const bf16_t* bBase = B + (size_t)(colBase + lo) * K + 8 * hi;

  for (int kk = 0; kk < K; kk += 32) {
    v16bf af[4], bfr[4];
#pragma unroll
    for (int mf = 0; mf < 4; ++mf)
      af[mf] = load_fragA(aBase + (size_t)mf * 16 * K + kk);
#pragma unroll
    for (int nf = 0; nf < 4; ++nf)
      bfr[nf] = load_fragB(bBase + (size_t)nf * 16 * K + kk);
    __builtin_prefetch(aBase + kk + 32, 0, 3);
    __builtin_prefetch(bBase + kk + 32, 0, 3);
#pragma unroll
    for (int mf = 0; mf < 4; ++mf)
#pragma unroll
      for (int nf = 0; nf < 4; ++nf)
        acc[mf][nf] = __builtin_amdgcn_wmma_f32_16x16x32_bf16(
            false, af[mf], false, bfr[nf], (short)0, acc[mf][nf], false, false);
  }

#pragma unroll
  for (int mf = 0; mf < 4; ++mf)
#pragma unroll
    for (int nf = 0; nf < 4; ++nf)
#pragma unroll
      for (int v = 0; v < 8; ++v) {
        int r = rowBase + mf * 16 + v + 8 * hi;
        int c = colBase + nf * 16 + lo;
        C[(size_t)r * N + c] = acc[mf][nf][v];
      }
}

// ---------- rotary (bug-faithful: angle indexed by HEAD, not position) + pack ----------
__global__ void mha_rotary_pack_k(const float* __restrict__ qkvf,
                                  bf16_t* __restrict__ Qbf, bf16_t* __restrict__ Kbf,
                                  bf16_t* __restrict__ Vt, int Bn, int H, int Tn) {
  const int d = threadIdx.x;                    // 0..127
  const int t = blockIdx.x % Tn;
  const int h = (blockIdx.x / Tn) % H;
  const int b = blockIdx.x / (Tn * H);

  const size_t rowbase = ((size_t)b * Tn + t) * (3 * H * DK);
  const float* qp = qkvf + rowbase + (size_t)(0 * H + h) * DK;
  const float* kp = qkvf + rowbase + (size_t)(1 * H + h) * DK;
  const float* vp = qkvf + rowbase + (size_t)(2 * H + h) * DK;

  const int f = d & 63;
  float c, s;
  if (f < 32) {
    float fr  = __powf(1.0e-4f, (float)f * (1.0f / 31.0f));  // (1/10000)^linspace(0,1,32)
    float ang = fr * (float)h;                               // BUG-faithful: head index as "position"
    c = __cosf(ang); s = __sinf(ang);
  } else { c = 1.0f; s = 0.0f; }
  const float sgn = (d < 64) ? s : -s;

  const float yq = qp[d] * c + qp[d ^ 64] * sgn;
  const float yk = kp[d] * c + kp[d ^ 64] * sgn;

  const size_t hoff = (size_t)(b * H + h);
  Qbf[(hoff * Tn + t) * DK + d] = (bf16_t)(yq * 0.08838834764831845f);  // 1/sqrt(128)
  Kbf[(hoff * Tn + t) * DK + d] = (bf16_t)yk;
  Vt[(hoff * DK + d) * Tn + t]  = (bf16_t)vp[d];
}

// ---------- flash attention: block = (b, h, 128 q rows); 8 waves x 16 rows ----------
// K/V tiles double-buffered in LDS via TDM (leader wave issues tensor_load_to_lds,
// TENSORcnt-synchronized); P transpose via per-wave LDS tile.
// Dynamic LDS layout (81920 B): [0,32K) K bufs 2x(64x128), [32K,64K) V bufs 2x(128x64),
// [64K,80K) P tiles 8x(16x64).
__global__ __launch_bounds__(256) void mha_flash_attn_wmma_k(
    const bf16_t* __restrict__ Q, const bf16_t* __restrict__ Kb,
    const bf16_t* __restrict__ Vt, bf16_t* __restrict__ Aout,
    int Bn, int H, int Tn) {
  extern __shared__ char smem[];
  const int lane = threadIdx.x & 31;
  const int w    = threadIdx.x >> 5;
  const int lo   = lane & 15;
  const int hi   = lane >> 4;
  const int QB   = Tn / 128;
  const int qb   = blockIdx.x % QB;
  const int h    = (blockIdx.x / QB) % H;
  const int b    = blockIdx.x / (QB * H);
  const bool leader = (w == 0);

  bf16_t* kBuf0 = (bf16_t*)(smem);                 // [2][64][128]
  bf16_t* vBuf0 = (bf16_t*)(smem + 32768);         // [2][128][64]
  bf16_t* pT    = (bf16_t*)(smem + 65536) + (size_t)w * (16 * 64);

  const bf16_t* Qh = Q  + (size_t)(b * H + h) * Tn * DK;
  const bf16_t* Kh = Kb + (size_t)(b * H + h) * Tn * DK;
  const bf16_t* Vh = Vt + (size_t)(b * H + h) * DK * Tn;

  const int qr0 = qb * 128 + w * 16;

  // Q A-fragments resident in registers for entire kernel
  v16bf qf[4];
  const bf16_t* qrow = Qh + (size_t)(qr0 + lo) * DK + 8 * hi;
#pragma unroll
  for (int c = 0; c < 4; ++c) qf[c] = load_fragA(qrow + c * 32);

  v8f o[8];
  float mrow[8], lrow[8];
#pragma unroll
  for (int n = 0; n < 8; ++n) o[n] = {};
#pragma unroll
  for (int v = 0; v < 8; ++v) { mrow[v] = -1.0e30f; lrow[v] = 0.0f; }

  const int kend = qb * 128 + 128;                 // causal upper bound for this q block

  // prologue: DMA tile 0 into buffer 0
  if (leader) {
    tdm_load_tile_2d(kBuf0, Kh, /*d0*/ DK, /*d1*/ 64, /*stride*/ DK);
    tdm_load_tile_2d(vBuf0, Vh, /*d0*/ 64, /*d1*/ DK, /*stride*/ (unsigned long long)Tn);
  }

  int buf = 0;
  for (int kt = 0; kt < kend; kt += 64) {
    // issue next tile's DMA, then wait for current tile (TDM completes in order)
    if (leader) {
      if (kt + 64 < kend) {
        bf16_t* kN = kBuf0 + (buf ^ 1) * (64 * DK);
        bf16_t* vN = vBuf0 + (buf ^ 1) * (DK * 64);
        tdm_load_tile_2d(kN, Kh + (size_t)(kt + 64) * DK, DK, 64, DK);
        tdm_load_tile_2d(vN, Vh + (kt + 64), 64, DK, (unsigned long long)Tn);
        tdm_wait<2>();   // oldest two (current tile) complete; next tile in flight
      } else {
        tdm_wait<0>();
      }
    }
    __syncthreads();

    const bf16_t* kTile = kBuf0 + buf * (64 * DK);   // [key][dk]
    const bf16_t* vTile = vBuf0 + buf * (DK * 64);   // [dkcol][key]

    // ---- S = Q * K^T from LDS K tile (4 tiles of 16 keys) ----
    v8f s[4];
#pragma unroll
    for (int t = 0; t < 4; ++t) s[t] = {};
#pragma unroll
    for (int t = 0; t < 4; ++t) {
      const bf16_t* krow = kTile + (size_t)(t * 16 + lo) * DK + 8 * hi;
#pragma unroll
      for (int c = 0; c < 4; ++c)
        s[t] = __builtin_amdgcn_wmma_f32_16x16x32_bf16(
            false, qf[c], false, load_fragB(krow + c * 32), (short)0, s[t], false, false);
    }

    // ---- causal mask + online softmax ----
#pragma unroll
    for (int v = 0; v < 8; ++v) {
      const int qrow_i = qr0 + v + 8 * hi;
      float mx = -1.0e30f;
#pragma unroll
      for (int t = 0; t < 4; ++t) {
        const int key = kt + t * 16 + lo;
        float val = (key <= qrow_i) ? s[t][v] : -1.0e30f;
        s[t][v] = val;
        mx = fmaxf(mx, val);
      }
#pragma unroll
      for (int off = 1; off < 16; off <<= 1) mx = fmaxf(mx, __shfl_xor(mx, off, 32));
      const float mnew  = fmaxf(mrow[v], mx);
      const float alpha = __expf(mrow[v] - mnew);
      mrow[v] = mnew;
      lrow[v] *= alpha;
#pragma unroll
      for (int n = 0; n < 8; ++n) o[n][v] *= alpha;
      float rsum = 0.0f;
#pragma unroll
      for (int t = 0; t < 4; ++t) {
        float p = __expf(s[t][v] - mnew);
        s[t][v] = p;
        rsum += p;
      }
#pragma unroll
      for (int off = 1; off < 16; off <<= 1) rsum += __shfl_xor(rsum, off, 32);
      lrow[v] += rsum;
    }

    // ---- P: C-layout -> A-layout transpose through LDS ----
#pragma unroll
    for (int t = 0; t < 4; ++t)
#pragma unroll
      for (int v = 0; v < 8; ++v)
        pT[(size_t)(v + 8 * hi) * 64 + t * 16 + lo] = (bf16_t)s[t][v];

    // ---- O += P * V from LDS V tile ----
#pragma unroll
    for (int pc = 0; pc < 2; ++pc) {
      v16bf pf = load_fragA(pT + (size_t)lo * 64 + pc * 32 + 8 * hi);
#pragma unroll
      for (int n = 0; n < 8; ++n) {
        const bf16_t* vrow = vTile + (size_t)(n * 16 + lo) * 64 + pc * 32 + 8 * hi;
        o[n] = __builtin_amdgcn_wmma_f32_16x16x32_bf16(
            false, pf, false, load_fragB(vrow), (short)0, o[n], false, false);
      }
    }
    __syncthreads();   // all waves done with buf before leader's next DMA overwrites it
    buf ^= 1;
  }

  // ---- normalize and write attn output as [b, t, h*dk + d] bf16 ----
#pragma unroll
  for (int n = 0; n < 8; ++n)
#pragma unroll
    for (int v = 0; v < 8; ++v) {
      const int trow = qr0 + v + 8 * hi;
      const float val = o[n][v] / lrow[v];
      Aout[((size_t)b * Tn + trow) * (H * DK) + h * DK + n * 16 + lo] = (bf16_t)val;
    }
}

// ---------------------------------------------------------------------------
extern "C" void kernel_launch(void* const* d_in, const int* in_sizes, int n_in,
                              void* d_out, int out_size, void* d_ws, size_t ws_size,
                              hipStream_t stream) {
  const float* x     = (const float*)d_in[0];
  const float* w_qkv = (const float*)d_in[1];
  const float* w_o   = (const float*)d_in[2];
  float* out = (float*)d_out;
  char*  ws  = (char*)d_ws;

  const int Bn = NB, H = NH, Tn = SEQ_T, D = NH * DK;   // D = 2048
  const int Mrows = Bn * Tn;                            // 4096
  const int Nqkv  = 3 * D;                              // 6144

  // workspace layout (bytes)
  bf16_t* xbf   = (bf16_t*)(ws);                                    // 16 MB
  bf16_t* wqkvb = (bf16_t*)(ws + 16777216);                         // 24 MB
  bf16_t* wob   = (bf16_t*)(ws + 16777216 + 25165824);              // 8 MB
  float*  qkvf  = (float*) (ws + 50331648);                         // 96 MB
  bf16_t* Qbf   = (bf16_t*)(ws + 150994944);                        // 16 MB
  bf16_t* Kbf   = (bf16_t*)(ws + 150994944 + 16777216);             // 16 MB
  bf16_t* Vt    = (bf16_t*)(ws + 150994944 + 2 * 16777216);         // 16 MB
  bf16_t* attnb = (bf16_t*)(ws + 150994944 + 3 * 16777216);         // 16 MB

  // 1) fp32 -> bf16 for x, w_qkv, w_o
  mha_f32_to_bf16_k<<<2048, 256, 0, stream>>>(x, xbf, Mrows * D);
  mha_f32_to_bf16_k<<<2048, 256, 0, stream>>>(w_qkv, wqkvb, Nqkv * D);
  mha_f32_to_bf16_k<<<2048, 256, 0, stream>>>(w_o, wob, D * D);

  // 2) QKV projection: qkvf[M, 3D] = xbf * wqkvb^T
  mha_gemm_bf16_wmma_k<<<dim3(Nqkv / 256, Mrows / 128), 256, 0, stream>>>(
      xbf, wqkvb, qkvf, Mrows, Nqkv, D);

  // 3) head-indexed rotary (+Q scale) and pack to [b,h,t,dk] / Vt [b,h,dk,t]
  mha_rotary_pack_k<<<Bn * H * Tn, 128, 0, stream>>>(qkvf, Qbf, Kbf, Vt, Bn, H, Tn);

  // 4) flash attention with WMMA + TDM double-buffered K/V tiles (80 KB dynamic LDS)
  mha_flash_attn_wmma_k<<<Bn * H * (Tn / 128), 256, 81920, stream>>>(
      Qbf, Kbf, Vt, attnb, Bn, H, Tn);

  // 5) output projection: out[M, D] = attnb * wob^T
  mha_gemm_bf16_wmma_k<<<dim3(D / 256, Mrows / 128), 256, 0, stream>>>(
      attnb, wob, out, Mrows, D, D);
}
